// Deformable_Conv_Block_80049600462926
// MI455X (gfx1250) — matrix operand order repaired and checked by reading, not compiled, not run on hardware
//
#include <hip/hip_runtime.h>
#include <math.h>

// ---------------------------------------------------------------------------
// Deformable conv block on gfx1250 (MI455X).
// All contractions via v_wmma_f32_16x16x32_f16 (wave32 WMMA), 2x2 accumulator
// tiles per wave, bilinear sampling in f32 VALU.
// ---------------------------------------------------------------------------

#define BDIM 256
static constexpr int Bn = 4;
static constexpr int Cc = 64;          // C
static constexpr int Hh = 160;
static constexpr int Ww = 160;
static constexpr int HW = Hh * Ww;     // 25600 (divisible by 128 -> tiles
                                       // never cross batch boundaries)
static constexpr int C2 = 128;         // 2*C
static constexpr int KT = 9;           // 3x3 taps

typedef _Float16     half16 __attribute__((ext_vector_type(16)));
typedef float        float8 __attribute__((ext_vector_type(8)));
typedef float        f32x4  __attribute__((ext_vector_type(4)));
typedef unsigned int u32x4  __attribute__((ext_vector_type(4)));

union FragH { half16 h; u32x4 q[2]; };

__device__ __forceinline__ int iclamp(int v, int lo, int hi) {
  return v < lo ? lo : (v > hi ? hi : v);
}

// ---------------------------------------------------------------------------
// Pack concat([lr, hr]) (f32 NCHW) -> f16 [B][128][HW]
// ---------------------------------------------------------------------------
__global__ void pack_concat_f16(const float* __restrict__ lr,
                                const float* __restrict__ hr,
                                _Float16* __restrict__ out) {
  int idx = blockIdx.x * BDIM + threadIdx.x;          // < B*C2*HW
  if (idx >= Bn * C2 * HW) return;
  int b   = idx / (C2 * HW);
  int r   = idx - b * (C2 * HW);
  int c   = r / HW;
  int pix = r - c * HW;
  float v = (c < Cc) ? lr[((size_t)b * Cc + c) * HW + pix]
                     : hr[((size_t)b * Cc + (c - Cc)) * HW + pix];
  out[idx] = (_Float16)v;
}

// ---------------------------------------------------------------------------
// Pack OIHW f32 weights -> f16 [O][k] with k = tap*Cin + cin (contiguous cin
// inside each tap so every 32-wide K slice is a coalesced contiguous load).
// ---------------------------------------------------------------------------
__global__ void pack_weight_f16(const float* __restrict__ w,
                                _Float16* __restrict__ out, int O, int Cin) {
  int total = O * Cin * KT;
  int idx = blockIdx.x * BDIM + threadIdx.x;
  if (idx >= total) return;
  int o   = idx / (Cin * KT);
  int r   = idx - o * (Cin * KT);
  int cin = r / KT;
  int tap = r - cin * KT;
  out[(size_t)o * (Cin * KT) + tap * Cin + cin] = (_Float16)w[idx];
}

// ---------------------------------------------------------------------------
// Implicit-GEMM 3x3 SAME conv:  M = B*HW pixels, N = Cout, K = 9*Cin.
// Block tile: 128 px x 64 ch; 8 waves arranged 4(M) x 2(N); each wave holds a
// 2x2 grid of 16x16 f32 accumulators (4 WMMAs per K-step from 2 A + 2 B
// fragments).
// MODE 0: +bias, relu, f16 output (hidden activations)
// MODE 1: raw f32 output (offset conv)
// ---------------------------------------------------------------------------
template <int MODE>
__global__ __launch_bounds__(BDIM)
void conv3x3_wmma(const _Float16* __restrict__ in,    // [B][Cin][HW] f16
                  const _Float16* __restrict__ wpk,   // [Cout][9*Cin] packed
                  const float* __restrict__ bias,
                  void* __restrict__ outv,
                  int Cin, int Cout) {
  // 80-byte row pitch: 16B-aligned for ds_load_b128 fragments, bank-friendly.
  __shared__ __align__(16) _Float16 lA[128 * 40];
  __shared__ __align__(16) _Float16 lB[64 * 40];

  const int t    = threadIdx.x;
  const int lane = t & 31;
  const int wave = t >> 5;
  const int wm   = wave & 3;           // M sub-tile (0..3) -> 32 px
  const int wn   = wave >> 2;          // N sub-tile (0..1) -> 32 ch
  const int half = lane >> 4;
  const int l15  = lane & 15;
  const int Ktot = Cin * KT;

  const int mtile = blockIdx.x;
  const int ntile = blockIdx.y;

  // Per-block constants: tile never crosses a batch boundary.
  const int bidx    = (mtile * 128) / HW;
  const int pixbase = mtile * 128 - bidx * HW;

  // A-staging role: one pixel, 16 contiguous k (channels) per thread
  const int pA  = t & 127;
  const int cA  = t >> 7;              // 0/1 -> k-halves [0,16) / [16,32)
  const int pix = pixbase + pA;
  const int yy  = pix / Ww;
  const int xx  = pix - yy * Ww;

  // B-staging role: one out-channel row, 8 contiguous k per thread
  const int nB = t >> 2;
  const int kB = (t & 3) * 8;
  const int ng = ntile * 64 + nB;

  float8 acc[2][2];
#pragma unroll
  for (int i = 0; i < 2; ++i)
#pragma unroll
    for (int j = 0; j < 2; ++j)
      acc[i][j] = (float8){0.f, 0.f, 0.f, 0.f, 0.f, 0.f, 0.f, 0.f};

  for (int kk = 0; kk < Ktot; kk += 32) {
    // ---- stage A: 128 px x 32 k (single tap per step since 32 | Cin) ----
    const int tap = kk / Cin;
    const int c0  = kk - tap * Cin;
    const int sy  = yy + tap / 3 - 1;
    const int sx  = xx + tap % 3 - 1;
    const bool inb = (sy >= 0) & (sy < Hh) & (sx >= 0) & (sx < Ww);
    const _Float16* src =
        in + ((size_t)bidx * Cin + c0 + cA * 16) * HW + sy * Ww + sx;
    __align__(16) _Float16 areg[16];
#pragma unroll
    for (int j = 0; j < 16; ++j)
      areg[j] = inb ? src[(size_t)j * HW] : (_Float16)0.f;
    *(u32x4*)&lA[pA * 40 + cA * 16]     = *(const u32x4*)&areg[0];
    *(u32x4*)&lA[pA * 40 + cA * 16 + 8] = *(const u32x4*)&areg[8];

    // ---- stage B: 64 n x 32 k (contiguous, coalesced) ----
    u32x4 bv = {0u, 0u, 0u, 0u};
    if (ng < Cout)
      bv = *(const u32x4*)(wpk + (size_t)ng * Ktot + kk + kB);
    *(u32x4*)&lB[nB * 40 + kB] = bv;
    if (ng < Cout && kk + 32 < Ktot)
      __builtin_prefetch(wpk + (size_t)ng * Ktot + kk + 32 + kB, 0, 0);
    __syncthreads();

    // ---- fragments per ISA 16-bit A 16x32 / B 32x16 layouts ----
    FragH a0, a1, b0, b1;
    const _Float16* ar0 = &lA[(wm * 32 + l15) * 40];
    const _Float16* ar1 = ar0 + 16 * 40;
    a0.q[0] = *(const u32x4*)(ar0 + half * 8);
    a0.q[1] = *(const u32x4*)(ar0 + 16 + half * 8);
    a1.q[0] = *(const u32x4*)(ar1 + half * 8);
    a1.q[1] = *(const u32x4*)(ar1 + 16 + half * 8);
    const _Float16* br0 = &lB[(wn * 32 + l15) * 40];
    const _Float16* br1 = br0 + 16 * 40;
    b0.q[0] = *(const u32x4*)(br0 + half * 8);
    b0.q[1] = *(const u32x4*)(br0 + 16 + half * 8);
    b1.q[0] = *(const u32x4*)(br1 + half * 8);
    b1.q[1] = *(const u32x4*)(br1 + 16 + half * 8);

    acc[0][0] = __builtin_amdgcn_wmma_f32_16x16x32_f16(
        false, a0.h, false, b0.h, (short)0, acc[0][0], false, false);
    acc[0][1] = __builtin_amdgcn_wmma_f32_16x16x32_f16(
        false, a0.h, false, b1.h, (short)0, acc[0][1], false, false);
    acc[1][0] = __builtin_amdgcn_wmma_f32_16x16x32_f16(
        false, a1.h, false, b0.h, (short)0, acc[1][0], false, false);
    acc[1][1] = __builtin_amdgcn_wmma_f32_16x16x32_f16(
        false, a1.h, false, b1.h, (short)0, acc[1][1], false, false);
    __syncthreads();
  }

  // ---- epilogue: C/D layout: VGPR r -> M = r + 8*half, N = l15 ----
  // Consecutive r are consecutive pixels -> 16B vector stores.
#pragma unroll
  for (int i = 0; i < 2; ++i) {
#pragma unroll
    for (int j = 0; j < 2; ++j) {
      const int n = ntile * 64 + wn * 32 + j * 16 + l15;
      if (n >= Cout) continue;
      const int op0 = pixbase + wm * 32 + i * 16 + half * 8;
      const float ba = (MODE == 0) ? bias[n] : 0.f;
      if (MODE == 0) {
        __align__(16) _Float16 o[8];
#pragma unroll
        for (int r = 0; r < 8; ++r) {
          float v = acc[i][j][r] + ba;
          o[r] = (_Float16)(v > 0.f ? v : 0.f);
        }
        *(u32x4*)((_Float16*)outv + ((size_t)bidx * Cout + n) * HW + op0) =
            *(const u32x4*)o;
      } else {
        float* po = (float*)outv + ((size_t)bidx * Cout + n) * HW + op0;
        f32x4 lo = {acc[i][j][0], acc[i][j][1], acc[i][j][2], acc[i][j][3]};
        f32x4 hi = {acc[i][j][4], acc[i][j][5], acc[i][j][6], acc[i][j][7]};
        *(f32x4*)po       = lo;
        *(f32x4*)(po + 4) = hi;
      }
    }
  }
}

// ---------------------------------------------------------------------------
// Deformable stage: bilinear-sample hr at 9 offset taps (f32), pack samples
// to f16 LDS A tile, contract with deform weights via WMMA.
//   M = B*HW, N = 64, K = 9*C = 576, k = tap*C + c.  Same 128x64 block tile.
// ---------------------------------------------------------------------------
__global__ __launch_bounds__(BDIM)
void deform_wmma(const float* __restrict__ hr,      // [B][C][HW] f32
                 const float* __restrict__ offs,    // [B][18][HW] f32
                 const _Float16* __restrict__ wpk,  // [64][576] packed
                 float* __restrict__ out) {         // [B][64][HW] f32
  __shared__ __align__(16) _Float16 lA[128 * 40];
  __shared__ __align__(16) _Float16 lB[64 * 40];

  const int t    = threadIdx.x;
  const int lane = t & 31;
  const int wave = t >> 5;
  const int wm   = wave & 3;
  const int wn   = wave >> 2;
  const int half = lane >> 4;
  const int l15  = lane & 15;
  const int Ktot = Cc * KT;            // 576

  const int mtile = blockIdx.x;

  const int bidx    = (mtile * 128) / HW;
  const int pixbase = mtile * 128 - bidx * HW;

  const int pA  = t & 127;
  const int cA  = t >> 7;
  const int pix = pixbase + pA;
  const int yy  = pix / Ww;
  const int xx  = pix - yy * Ww;

  const int nB = t >> 2;
  const int kB = (t & 3) * 8;
  const int ng = nB;                   // N = 64 exactly, single n-tile

  float8 acc[2][2];
#pragma unroll
  for (int i = 0; i < 2; ++i)
#pragma unroll
    for (int j = 0; j < 2; ++j)
      acc[i][j] = (float8){0.f, 0.f, 0.f, 0.f, 0.f, 0.f, 0.f, 0.f};

  for (int kk = 0; kk < Ktot; kk += 32) {
    const int tap = kk / Cc;           // 0..8 (two 32-steps per tap)
    const int c0  = kk - tap * Cc;     // 0 or 32
    // --- bilinear parameters for (pixel, tap), f32 like the reference ---
    const float dy = offs[((size_t)bidx * 18 + tap * 2) * HW + pix];
    const float dx = offs[((size_t)bidx * 18 + tap * 2 + 1) * HW + pix];
    const float py = (float)(yy + tap / 3 - 1) + dy;
    const float px = (float)(xx + tap % 3 - 1) + dx;
    const float fy = floorf(py), fx = floorf(px);
    const float wy = py - fy, wx = px - fx;
    const int y0 = (int)fy, x0 = (int)fx;
    const int y1 = y0 + 1, x1 = x0 + 1;
    const float v00 = (y0 >= 0 && y0 < Hh && x0 >= 0 && x0 < Ww) ? 1.f : 0.f;
    const float v01 = (y0 >= 0 && y0 < Hh && x1 >= 0 && x1 < Ww) ? 1.f : 0.f;
    const float v10 = (y1 >= 0 && y1 < Hh && x0 >= 0 && x0 < Ww) ? 1.f : 0.f;
    const float v11 = (y1 >= 0 && y1 < Hh && x1 >= 0 && x1 < Ww) ? 1.f : 0.f;
    const int yc0 = iclamp(y0, 0, Hh - 1), yc1 = iclamp(y1, 0, Hh - 1);
    const int xc0 = iclamp(x0, 0, Ww - 1), xc1 = iclamp(x1, 0, Ww - 1);
    const int a00 = yc0 * Ww + xc0, a01 = yc0 * Ww + xc1;
    const int a10 = yc1 * Ww + xc0, a11 = yc1 * Ww + xc1;
    const float w00 = (1.f - wy) * (1.f - wx) * v00;
    const float w01 = (1.f - wy) * wx * v01;
    const float w10 = wy * (1.f - wx) * v10;
    const float w11 = wy * wx * v11;

    const float* hp = hr + ((size_t)bidx * Cc + c0 + cA * 16) * HW;
    __align__(16) _Float16 areg[16];
#pragma unroll
    for (int j = 0; j < 16; ++j) {
      const float* p = hp + (size_t)j * HW;
      float s = w00 * p[a00] + w01 * p[a01] + w10 * p[a10] + w11 * p[a11];
      areg[j] = (_Float16)s;
    }
    *(u32x4*)&lA[pA * 40 + cA * 16]     = *(const u32x4*)&areg[0];
    *(u32x4*)&lA[pA * 40 + cA * 16 + 8] = *(const u32x4*)&areg[8];

    *(u32x4*)&lB[nB * 40 + kB] =
        *(const u32x4*)(wpk + (size_t)ng * Ktot + kk + kB);
    __syncthreads();

    FragH a0, a1, b0, b1;
    const _Float16* ar0 = &lA[(wm * 32 + l15) * 40];
    const _Float16* ar1 = ar0 + 16 * 40;
    a0.q[0] = *(const u32x4*)(ar0 + half * 8);
    a0.q[1] = *(const u32x4*)(ar0 + 16 + half * 8);
    a1.q[0] = *(const u32x4*)(ar1 + half * 8);
    a1.q[1] = *(const u32x4*)(ar1 + 16 + half * 8);
    const _Float16* br0 = &lB[(wn * 32 + l15) * 40];
    const _Float16* br1 = br0 + 16 * 40;
    b0.q[0] = *(const u32x4*)(br0 + half * 8);
    b0.q[1] = *(const u32x4*)(br0 + 16 + half * 8);
    b1.q[0] = *(const u32x4*)(br1 + half * 8);
    b1.q[1] = *(const u32x4*)(br1 + 16 + half * 8);

    acc[0][0] = __builtin_amdgcn_wmma_f32_16x16x32_f16(
        false, a0.h, false, b0.h, (short)0, acc[0][0], false, false);
    acc[0][1] = __builtin_amdgcn_wmma_f32_16x16x32_f16(
        false, a0.h, false, b1.h, (short)0, acc[0][1], false, false);
    acc[1][0] = __builtin_amdgcn_wmma_f32_16x16x32_f16(
        false, a1.h, false, b0.h, (short)0, acc[1][0], false, false);
    acc[1][1] = __builtin_amdgcn_wmma_f32_16x16x32_f16(
        false, a1.h, false, b1.h, (short)0, acc[1][1], false, false);
    __syncthreads();
  }

#pragma unroll
  for (int i = 0; i < 2; ++i) {
#pragma unroll
    for (int j = 0; j < 2; ++j) {
      const int n = wn * 32 + j * 16 + l15;           // < 64 always
      const int op0 = pixbase + wm * 32 + i * 16 + half * 8;
      float* po = out + ((size_t)bidx * Cc + n) * HW + op0;
      f32x4 lo = {acc[i][j][0], acc[i][j][1], acc[i][j][2], acc[i][j][3]};
      f32x4 hi = {acc[i][j][4], acc[i][j][5], acc[i][j][6], acc[i][j][7]};
      *(f32x4*)po       = lo;
      *(f32x4*)(po + 4) = hi;
    }
  }
}

// ---------------------------------------------------------------------------
// Workspace layout (bytes)
// ---------------------------------------------------------------------------
static constexpr size_t OFF_X   = 0;                         // f16 [4][128][HW]
static constexpr size_t OFF_H1  = 26214400;                  // f16 [4][128][HW]
static constexpr size_t OFF_OFF = 52428800;                  // f32 [4][18][HW]
static constexpr size_t OFF_W1  = 59801600;                  // f16 [128][1152]
static constexpr size_t OFF_W2  = 60096512;                  // f16 [128][1152]
static constexpr size_t OFF_WO  = 60391424;                  // f16 [18][1152]
static constexpr size_t OFF_WD  = 60432896;                  // f16 [64][576]
static constexpr size_t WS_NEED = 60506624;

extern "C" void kernel_launch(void* const* d_in, const int* in_sizes, int n_in,
                              void* d_out, int out_size, void* d_ws,
                              size_t ws_size, hipStream_t stream) {
  (void)in_sizes; (void)n_in; (void)out_size;
  if (ws_size < WS_NEED) return;

  const float* lr = (const float*)d_in[0];
  const float* hr = (const float*)d_in[1];
  const float* w1 = (const float*)d_in[2];
  const float* b1 = (const float*)d_in[3];
  const float* w2 = (const float*)d_in[4];
  const float* b2 = (const float*)d_in[5];
  const float* wo = (const float*)d_in[6];
  const float* wd = (const float*)d_in[7];
  float* out = (float*)d_out;
  char* ws = (char*)d_ws;

  _Float16* xbuf = (_Float16*)(ws + OFF_X);
  _Float16* h1b  = (_Float16*)(ws + OFF_H1);
  float*    offb = (float*)(ws + OFF_OFF);
  _Float16* w1pk = (_Float16*)(ws + OFF_W1);
  _Float16* w2pk = (_Float16*)(ws + OFF_W2);
  _Float16* wopk = (_Float16*)(ws + OFF_WO);
  _Float16* wdpk = (_Float16*)(ws + OFF_WD);

  // 1) precision/layout packing
  pack_concat_f16<<<(Bn * C2 * HW + BDIM - 1) / BDIM, BDIM, 0, stream>>>(lr, hr, xbuf);
  pack_weight_f16<<<(C2 * C2 * KT + BDIM - 1) / BDIM, BDIM, 0, stream>>>(w1, w1pk, C2, C2);
  pack_weight_f16<<<(C2 * C2 * KT + BDIM - 1) / BDIM, BDIM, 0, stream>>>(w2, w2pk, C2, C2);
  pack_weight_f16<<<(2 * KT * C2 * KT + BDIM - 1) / BDIM, BDIM, 0, stream>>>(wo, wopk, 2 * KT, C2);
  pack_weight_f16<<<(Cc * Cc * KT + BDIM - 1) / BDIM, BDIM, 0, stream>>>(wd, wdpk, Cc, Cc);

  const int mtiles = (Bn * HW) / 128;  // 800

  // 2) conv1: x -> h1 (relu+bias, f16)
  conv3x3_wmma<0><<<dim3(mtiles, 2), BDIM, 0, stream>>>(xbuf, w1pk, b1, (void*)h1b, C2, C2);
  // 3) conv2: h1 -> h2 (reuse x buffer)
  conv3x3_wmma<0><<<dim3(mtiles, 2), BDIM, 0, stream>>>(h1b, w2pk, b2, (void*)xbuf, C2, C2);
  // 4) offset conv: h2 -> offsets (raw f32, Cout=18 in one 64-wide n-tile)
  conv3x3_wmma<1><<<dim3(mtiles, 1), BDIM, 0, stream>>>(xbuf, wopk, nullptr, (void*)offb, C2, 2 * KT);
  // 5) deformable sampling + contraction -> d_out
  deform_wmma<<<dim3(mtiles, 1), BDIM, 0, stream>>>(hr, offb, wdpk, out);
}
